// AutoregressiveFeedback_8065948582510
// MI455X (gfx1250) — compile-verified, matrix operand order checked by reading, not run on hardware
//
#include <hip/hip_runtime.h>
#include <hip/hip_bf16.h>

// ---------------- problem dims ----------------
#define Bb 256
#define Tt 256
#define FAa 64
#define Uu 256
#define Hh 4
#define KDd 64
#define Ss 64
#define NFf 64

typedef __attribute__((ext_vector_type(16))) _Float16 v16h;
typedef __attribute__((ext_vector_type(8)))  _Float16 v8h;
typedef __attribute__((ext_vector_type(8)))  float    v8f;

__device__ inline v8f vzero() {
    v8f a;
#pragma unroll
    for (int i = 0; i < 8; ++i) a[i] = 0.f;
    return a;
}

// CDNA5 WMMA f32 = f16 x f16 + f32, 16x16x32
__device__ inline v8f wmma32(v16h a, v16h b, v8f c) {
    return __builtin_amdgcn_wmma_f32_16x16x32_f16(false, a, false, b, (short)0, c, false, false);
}

// Load a 16x32 f16 fragment from a row-major matrix (A-frag; also used for
// B-frags from [N,K]-transposed weights). Per CDNA5 ISA layout:
// lanes 0-15: row=lane,   K={0..7} in v0-3, K={16..23} in v4-7
// lanes16-31: row=lane-16,K={8..15},{24..31}
__device__ inline v16h ldfrag(const _Float16* p, int ld, int lane) {
    int m  = lane & 15;
    int kb = (lane >> 4) << 3;  // 0 or 8
    const _Float16* q = p + (size_t)m * ld + kb;
    v8h lo = *reinterpret_cast<const v8h*>(q);
    v8h hi = *reinterpret_cast<const v8h*>(q + 16);
    v16h r;
#pragma unroll
    for (int i = 0; i < 8; ++i) { r[i] = lo[i]; r[8 + i] = hi[i]; }
    return r;
}

// C layout: element (m,n): n = lane&15, m = (lane>=16?8:0)+vgpr
__device__ inline void st_f32(float* C, int ldc, int lane, v8f c) {
    int n = lane & 15, mb = (lane >> 4) << 3;
#pragma unroll
    for (int r = 0; r < 8; ++r) C[(size_t)(mb + r) * ldc + n] = c[r];
}
__device__ inline void st_f16(_Float16* C, int ldc, int lane, v8f c) {
    int n = lane & 15, mb = (lane >> 4) << 3;
#pragma unroll
    for (int r = 0; r < 8; ++r) C[(size_t)(mb + r) * ldc + n] = (_Float16)c[r];
}
// Transposed store CT[n][m]: lane's 8 values are contiguous -> one b128 LDS store
__device__ inline void st_f16_T(_Float16* C, int ldc, int lane, v8f c) {
    int n = lane & 15, mb = (lane >> 4) << 3;
    v8h pk;
#pragma unroll
    for (int r = 0; r < 8; ++r) pk[r] = (_Float16)c[r];
    *reinterpret_cast<v8h*>(C + (size_t)n * ldc + mb) = pk;
}
// Transposed f32 store: one 32B vector store per lane
__device__ inline void st_f32_T(float* C, int ldc, int lane, v8f c) {
    int n = lane & 15, mb = (lane >> 4) << 3;
    *reinterpret_cast<v8f*>(C + (size_t)n * ldc + mb) = c;
}

__device__ inline float sigm(float x) { return 1.f / (1.f + __expf(-x)); }

// ---------------- conversion kernels ----------------
__global__ void k_cvt(const float* __restrict__ x, _Float16* __restrict__ y, int n) {
    int i = blockIdx.x * 256 + threadIdx.x;
    if (i < n) y[i] = (_Float16)x[i];
}
// w[K][N] f32 -> wt[N][K] f16
__global__ void k_cvtT(const float* __restrict__ w, _Float16* __restrict__ wt, int K, int N) {
    int i = blockIdx.x * 256 + threadIdx.x;
    if (i < K * N) {
        int k = i / N, n = i % N;
        wt[(size_t)n * K + k] = (_Float16)w[i];
    }
}

// ---------------- fused multi-head attention, one block per (b,h) ----------------
__global__ void __launch_bounds__(256, 1)
k_attn(const _Float16* __restrict__ inph,                // [B,T,FA] f16
       const _Float16* __restrict__ WqT, const float* __restrict__ bq,
       const _Float16* __restrict__ WkT, const float* __restrict__ bk,
       const _Float16* __restrict__ WvT, const float* __restrict__ bv,
       _Float16* __restrict__ attn)                      // [B,T,H*KD] f16
{
    extern __shared__ char sm[];
    _Float16* qs   = (_Float16*)sm;          // [256][64]
    _Float16* ks   = qs + 256 * 64;          // [256][64]
    _Float16* vT   = ks + 256 * 64;          // [64][256]
    _Float16* sbuf = vT + 64 * 256;          // 8 waves x [16][256]

    int tid = threadIdx.x, lane = tid & 31, wv = tid >> 5;
    int b = blockIdx.x >> 2, h = blockIdx.x & 3;
    const _Float16* Xb = inph + (size_t)b * 256 * 64;

    // phase 1: q,k,v projections fused per tile: load A-frags once, use x3
    for (int j = 0; j < 8; ++j) {
        int rem = wv + j * 8, mt = rem >> 2, nt = rem & 3;
        const _Float16* Ap = Xb + (size_t)mt * 16 * 64;
        v16h a0 = ldfrag(Ap, 64, lane);
        v16h a1 = ldfrag(Ap + 32, 64, lane);
        int col = h * 64 + nt * 16;
        {
            const _Float16* Bp = WqT + (size_t)col * 64;
            v8f a = vzero();
            a = wmma32(a0, ldfrag(Bp, 64, lane), a);
            a = wmma32(a1, ldfrag(Bp + 32, 64, lane), a);
            float bb = bq[col + (lane & 15)];
#pragma unroll
            for (int r = 0; r < 8; ++r) a[r] += bb;
            st_f16(qs + (size_t)mt * 16 * 64 + nt * 16, 64, lane, a);
        }
        {
            const _Float16* Bp = WkT + (size_t)col * 64;
            v8f a = vzero();
            a = wmma32(a0, ldfrag(Bp, 64, lane), a);
            a = wmma32(a1, ldfrag(Bp + 32, 64, lane), a);
            float bb = bk[col + (lane & 15)];
#pragma unroll
            for (int r = 0; r < 8; ++r) a[r] += bb;
            st_f16(ks + (size_t)mt * 16 * 64 + nt * 16, 64, lane, a);
        }
        {
            const _Float16* Bp = WvT + (size_t)col * 64;
            v8f a = vzero();
            a = wmma32(a0, ldfrag(Bp, 64, lane), a);
            a = wmma32(a1, ldfrag(Bp + 32, 64, lane), a);
            float bb = bv[col + (lane & 15)];
#pragma unroll
            for (int r = 0; r < 8; ++r) a[r] += bb;
            st_f16_T(vT + (size_t)nt * 16 * 256 + mt * 16, 256, lane, a);  // vectorized
        }
    }
    __syncthreads();

    // phase 2: per wave, two 16-row q tiles: scores -> register softmax -> @v
    _Float16* sw = sbuf + (size_t)wv * 16 * 256;
    const float sc = 0.125f;  // 1/sqrt(KD)
    for (int half = 0; half < 2; ++half) {
        int mt = wv + half * 8;
        v16h qa[2];
#pragma unroll
        for (int i = 0; i < 2; ++i) qa[i] = ldfrag(qs + (size_t)mt * 16 * 64 + i * 32, 64, lane);
        v8f acc[16];
#pragma unroll
        for (int nt = 0; nt < 16; ++nt) acc[nt] = vzero();
#pragma unroll
        for (int nt = 0; nt < 16; ++nt)
#pragma unroll
            for (int i = 0; i < 2; ++i)
                acc[nt] = wmma32(qa[i], ldfrag(ks + (size_t)nt * 16 * 64 + i * 32, 64, lane), acc[nt]);
        // softmax over the 256 columns of each of this wave's 16 rows
#pragma unroll
        for (int r = 0; r < 8; ++r) {
            float mx = -1e30f;
#pragma unroll
            for (int nt = 0; nt < 16; ++nt) mx = fmaxf(mx, acc[nt][r]);
            for (int d = 1; d < 16; d <<= 1) mx = fmaxf(mx, __shfl_xor(mx, d, 32));
            float ssum = 0.f;
#pragma unroll
            for (int nt = 0; nt < 16; ++nt) {
                float e = __expf((acc[nt][r] - mx) * sc);
                acc[nt][r] = e; ssum += e;
            }
            for (int d = 1; d < 16; d <<= 1) ssum += __shfl_xor(ssum, d, 32);
            float inv = 1.f / ssum;
#pragma unroll
            for (int nt = 0; nt < 16; ++nt) acc[nt][r] *= inv;
        }
#pragma unroll
        for (int nt = 0; nt < 16; ++nt) st_f16(sw + nt * 16, 256, lane, acc[nt]);
        // out[16,64] = softmax(s) @ v : hoist the 8 A-frags (acc is dead by now)
        v16h sa[8];
#pragma unroll
        for (int i = 0; i < 8; ++i) sa[i] = ldfrag(sw + i * 32, 256, lane);
        for (int nt2 = 0; nt2 < 4; ++nt2) {
            v8f a = vzero();
#pragma unroll
            for (int i = 0; i < 8; ++i)
                a = wmma32(sa[i], ldfrag(vT + (size_t)nt2 * 16 * 256 + i * 32, 256, lane), a);
            _Float16* op = attn + ((size_t)(b * 256 + mt * 16)) * 256 + h * 64 + nt2 * 16;
            st_f16(op, 256, lane, a);
        }
    }
}

// ---------------- x = attn @ Wo + bo  -> f16 [B*T,64] ----------------
__global__ void __launch_bounds__(256, 1)
k_proj(const _Float16* __restrict__ attn, const _Float16* __restrict__ WoT,
       const float* __restrict__ bo, _Float16* __restrict__ xh)
{
    int lane = threadIdx.x & 31, wv = threadIdx.x >> 5;
    int mt = blockIdx.x * 8 + wv;  // 4096 M-tiles
    const _Float16* A = attn + (size_t)mt * 16 * 256;
    v16h af[8];
#pragma unroll
    for (int i = 0; i < 8; ++i) af[i] = ldfrag(A + i * 32, 256, lane);
    for (int nt = 0; nt < 4; ++nt) {
        const _Float16* Bp = WoT + (size_t)nt * 16 * 256;
        __builtin_prefetch(Bp, 0, 1);
        v8f a = vzero();
#pragma unroll
        for (int i = 0; i < 8; ++i)
            a = wmma32(af[i], ldfrag(Bp + i * 32, 256, lane), a);
        float bb = bo[nt * 16 + (lane & 15)];
#pragma unroll
        for (int r = 0; r < 8; ++r) a[r] += bb;
        st_f16(xh + (size_t)mt * 16 * 64 + nt * 16, 64, lane, a);
    }
}

// zT[1024][16] = (X[16,KX] @ W^T + H[16,256] @ U^T)^T   (per-wave: 8 N-tiles)
// Only the h A-fragments (8 v16h = 64 VGPRs) are hoisted; x is hoisted just
// in the cheap KX=64 case. Keeps VGPR pressure below the spill threshold.
template <int KX>
__device__ inline void gemm_z_dual(const _Float16* X, int ldx, const _Float16* WT,
                                   const _Float16* Hs, const _Float16* UT,
                                   float* zT, int lane, int wv)
{
    v16h ah[8];
#pragma unroll
    for (int i = 0; i < 8; ++i) ah[i] = ldfrag(Hs + i * 32, 256, lane);
    v16h ax0, ax1;
    if (KX == 64) {
        ax0 = ldfrag(X, ldx, lane);
        ax1 = ldfrag(X + 32, ldx, lane);
    }
    for (int j = 0; j < 8; ++j) {
        int nt = wv * 8 + j;
        const _Float16* Bx = WT + (size_t)nt * 16 * KX;
        const _Float16* Bu = UT + (size_t)nt * 16 * 256;
        __builtin_prefetch(Bx, 0, 1);
        __builtin_prefetch(Bu, 0, 1);
        v8f a = vzero();
        if (KX == 64) {
            a = wmma32(ax0, ldfrag(Bx, 64, lane), a);
            a = wmma32(ax1, ldfrag(Bx + 32, 64, lane), a);
        } else {
#pragma unroll
            for (int i = 0; i < KX / 32; ++i)
                a = wmma32(ldfrag(X + i * 32, ldx, lane), ldfrag(Bx + i * 32, KX, lane), a);
        }
#pragma unroll
        for (int i = 0; i < 8; ++i)
            a = wmma32(ah[i], ldfrag(Bu + i * 32, 256, lane), a);
        st_f32_T(zT + (size_t)nt * 16 * 16, 16, lane, a);   // vectorized 32B store
    }
}

__device__ inline void lstm_gates(const float* zT, const float* bias,
                                  _Float16* h, float* c, int tid) {
    for (int idx = tid * 16; idx < tid * 16 + 16; ++idx) {
        int row = idx >> 8, u = idx & 255;
        float zi = zT[(size_t)(u)*16 + row]        + bias[u];
        float zf = zT[(size_t)(256 + u) * 16 + row] + bias[256 + u];
        float zg = zT[(size_t)(512 + u) * 16 + row] + bias[512 + u];
        float zo = zT[(size_t)(768 + u) * 16 + row] + bias[768 + u];
        float cn = sigm(zf) * c[row * 256 + u] + sigm(zi) * tanhf(zg);
        c[row * 256 + u] = cn;
        h[row * 256 + u] = (_Float16)(sigm(zo) * tanhf(cn));
    }
}

// feats chain: hin[16,256] -> t0 -> t1 -> xf[16,64]
__device__ inline void feats(const _Float16* hin, _Float16* t0, _Float16* t1, _Float16* xf,
                             const _Float16* F0T, const float* Fb0,
                             const _Float16* F1T, const float* Fb1,
                             const _Float16* F2T, const float* Fb2, int lane, int wv)
{
    {
        v16h ha[8];
#pragma unroll
        for (int i = 0; i < 8; ++i) ha[i] = ldfrag(hin + i * 32, 256, lane);
        for (int j = 0; j < 2; ++j) {
            int nt = wv * 2 + j;
            const _Float16* Bp = F0T + (size_t)nt * 16 * 256;
            v8f a = vzero();
#pragma unroll
            for (int i = 0; i < 8; ++i) a = wmma32(ha[i], ldfrag(Bp + i * 32, 256, lane), a);
            float bb = Fb0[nt * 16 + (lane & 15)];
#pragma unroll
            for (int r = 0; r < 8; ++r) a[r] += bb;
            st_f16(t0 + nt * 16, 256, lane, a);
        }
    }
    __syncthreads();
    {
        v16h ta[8];
#pragma unroll
        for (int i = 0; i < 8; ++i) ta[i] = ldfrag(t0 + i * 32, 256, lane);
        for (int j = 0; j < 2; ++j) {
            int nt = wv * 2 + j;
            const _Float16* Bp = F1T + (size_t)nt * 16 * 256;
            v8f a = vzero();
#pragma unroll
            for (int i = 0; i < 8; ++i) a = wmma32(ta[i], ldfrag(Bp + i * 32, 256, lane), a);
            float bb = Fb1[nt * 16 + (lane & 15)];
#pragma unroll
            for (int r = 0; r < 8; ++r) a[r] += bb;
            st_f16(t1 + nt * 16, 256, lane, a);
        }
    }
    __syncthreads();
    if (wv < 4) {
        int nt = wv;
        const _Float16* Bp = F2T + (size_t)nt * 16 * 256;
        v8f a = vzero();
#pragma unroll
        for (int k = 0; k < 256; k += 32)
            a = wmma32(ldfrag(t1 + k, 256, lane), ldfrag(Bp + k, 256, lane), a);
        float bb = Fb2[nt * 16 + (lane & 15)];
#pragma unroll
        for (int r = 0; r < 8; ++r) a[r] += bb;
        st_f16(xf + nt * 16, 64, lane, a);
    }
    __syncthreads();
}

__device__ inline void write_p(_Float16* p, const _Float16* h, int b0r, int s, int tid) {
    for (int i = tid; i < 16 * 256; i += 256) {
        int m = i >> 8, u = i & 255;
        p[((size_t)(b0r + m) * Ss + s) * Uu + u] = h[m * 256 + u];
    }
}

// ---------------- fused 3-layer LSTM warmup + autoregressive decode ----------------
// grid: 16 blocks x 256 threads; each block owns 16 batch rows; h/c persist in LDS.
__global__ void __launch_bounds__(256, 1)
k_lstm(const _Float16* __restrict__ xh,  // [B,T,64] f16
       const _Float16* __restrict__ W0T, const _Float16* __restrict__ U0T, const float* __restrict__ b0,
       const _Float16* __restrict__ W1T, const _Float16* __restrict__ U1T, const float* __restrict__ b1,
       const _Float16* __restrict__ W2T, const _Float16* __restrict__ U2T, const float* __restrict__ b2,
       const _Float16* __restrict__ F0T, const float* __restrict__ Fb0,
       const _Float16* __restrict__ F1T, const float* __restrict__ Fb1,
       const _Float16* __restrict__ F2T, const float* __restrict__ Fb2,
       _Float16* __restrict__ p)          // [B,S,U] f16
{
    extern __shared__ char sm[];
    float*    z  = (float*)sm;                       // zT: [1024][16] f32 (64KB)
    _Float16* h0 = (_Float16*)(sm + 65536);
    _Float16* h1 = h0 + 16 * 256;
    _Float16* h2 = h1 + 16 * 256;
    float*    c0 = (float*)(h2 + 16 * 256);
    float*    c1 = c0 + 16 * 256;
    float*    c2 = c1 + 16 * 256;
    _Float16* xf = (_Float16*)(c2 + 16 * 256);       // 16x64
    _Float16* t0 = xf + 16 * 64;                     // 16x256
    _Float16* t1 = t0 + 16 * 256;                    // 16x256

    int tid = threadIdx.x, lane = tid & 31, wv = tid >> 5;
    int b0r = blockIdx.x * 16;

    for (int i = tid; i < 16 * 256; i += 256) {
        h0[i] = (_Float16)0.f; h1[i] = (_Float16)0.f; h2[i] = (_Float16)0.f;
        c0[i] = 0.f; c1[i] = 0.f; c2[i] = 0.f;
    }
    __syncthreads();

    // --- warmup over T timesteps, all 3 layers fused ---
    for (int t = 0; t < Tt; ++t) {
        const _Float16* xt = xh + (size_t)b0r * (Tt * 64) + (size_t)t * 64;  // row stride T*64
        gemm_z_dual<64>(xt, Tt * 64, W0T, h0, U0T, z, lane, wv);
        __syncthreads();
        lstm_gates(z, b0, h0, c0, tid);
        __syncthreads();
        gemm_z_dual<256>(h0, 256, W1T, h1, U1T, z, lane, wv);
        __syncthreads();
        lstm_gates(z, b1, h1, c1, tid);
        __syncthreads();
        gemm_z_dual<256>(h1, 256, W2T, h2, U2T, z, lane, wv);
        __syncthreads();
        lstm_gates(z, b2, h2, c2, tid);
        __syncthreads();
    }

    // pred0 = h2
    write_p(p, h2, b0r, 0, tid);
    __syncthreads();
    feats(h2, t0, t1, xf, F0T, Fb0, F1T, Fb1, F2T, Fb2, lane, wv);

    // --- autoregressive decode: S-1 steps over lstm cells 0,1 ---
    for (int s = 1; s < Ss; ++s) {
        gemm_z_dual<64>(xf, 64, W0T, h0, U0T, z, lane, wv);
        __syncthreads();
        lstm_gates(z, b0, h0, c0, tid);
        __syncthreads();
        gemm_z_dual<256>(h0, 256, W1T, h1, U1T, z, lane, wv);
        __syncthreads();
        lstm_gates(z, b1, h1, c1, tid);
        __syncthreads();
        write_p(p, h1, b0r, s, tid);
        __syncthreads();
        feats(h1, t0, t1, xf, F0T, Fb0, F1T, Fb1, F2T, Fb2, lane, wv);
    }
}

// ---------------- final scaled self-attention + prediction head, one block per b ----
__global__ void __launch_bounds__(256, 1)
k_final(const _Float16* __restrict__ p, const float* __restrict__ attn_scale,
        const _Float16* __restrict__ predWT, const float* __restrict__ pred_b,
        float* __restrict__ out)
{
    extern __shared__ char sm[];
    _Float16* ps = (_Float16*)sm;            // [64][256]
    _Float16* pT = ps + 64 * 256;            // [256][64]
    float*    w  = (float*)(pT + 256 * 64);  // [64][64] f32
    _Float16* wh = (_Float16*)(w + 64 * 64); // [64][64] f16
    _Float16* o1 = wh + 64 * 64;             // [64][256]

    int b = blockIdx.x, tid = threadIdx.x, lane = tid & 31, wv = tid >> 5;
    for (int i = tid; i < 64 * 256; i += 256) {
        int s = i >> 8, u = i & 255;
        _Float16 v = p[((size_t)b * Ss + s) * Uu + u];
        ps[s * 256 + u] = v;
        pT[u * 64 + s]  = v;
    }
    __syncthreads();
    float scale = *attn_scale;

    // scores = scale * p @ p^T   (16 tiles; both of a wave's tiles share mt -> hoist A)
    {
        int mt = (wv * 2) >> 2;
        v16h pa[8];
#pragma unroll
        for (int i = 0; i < 8; ++i) pa[i] = ldfrag(ps + (size_t)mt * 16 * 256 + i * 32, 256, lane);
        for (int j = 0; j < 2; ++j) {
            int tt = wv * 2 + j, nt = tt & 3;
            v8f a = vzero();
#pragma unroll
            for (int i = 0; i < 8; ++i)
                a = wmma32(pa[i], ldfrag(ps + (size_t)nt * 16 * 256 + i * 32, 256, lane), a);
#pragma unroll
            for (int r = 0; r < 8; ++r) a[r] *= scale;
            st_f32(w + (size_t)mt * 16 * 64 + nt * 16, 64, lane, a);
        }
    }
    __syncthreads();
    if (tid < 64) {
        float mx = -1e30f;
        for (int jx = 0; jx < 64; ++jx) mx = fmaxf(mx, w[tid * 64 + jx]);
        float ssum = 0.f;
        for (int jx = 0; jx < 64; ++jx) { float e = __expf(w[tid * 64 + jx] - mx); w[tid * 64 + jx] = e; ssum += e; }
        float inv = 1.f / ssum;
        for (int jx = 0; jx < 64; ++jx) wh[tid * 64 + jx] = (_Float16)(w[tid * 64 + jx] * inv);
    }
    __syncthreads();
    // o1 = softmax(w) @ p   (64 tiles; per-wave mt constant -> hoist A)
    {
        int mt = (wv * 8) >> 4;
        v16h wa[2];
#pragma unroll
        for (int i = 0; i < 2; ++i) wa[i] = ldfrag(wh + (size_t)mt * 16 * 64 + i * 32, 64, lane);
        for (int j = 0; j < 8; ++j) {
            int tt = wv * 8 + j, nt = tt & 15;
            v8f a = vzero();
#pragma unroll
            for (int i = 0; i < 2; ++i)
                a = wmma32(wa[i], ldfrag(pT + (size_t)nt * 16 * 64 + i * 32, 64, lane), a);
            st_f16(o1 + (size_t)mt * 16 * 256 + nt * 16, 256, lane, a);
        }
    }
    __syncthreads();
    // out = o1 @ pred_W + pred_b  -> f32 [S,NF]
    {
        int mt = (wv * 2) >> 2;
        v16h oa[8];
#pragma unroll
        for (int i = 0; i < 8; ++i) oa[i] = ldfrag(o1 + (size_t)mt * 16 * 256 + i * 32, 256, lane);
        for (int j = 0; j < 2; ++j) {
            int tt = wv * 2 + j, nt = tt & 3;
            const _Float16* Bp = predWT + (size_t)nt * 16 * 256;
            v8f a = vzero();
#pragma unroll
            for (int i = 0; i < 8; ++i)
                a = wmma32(oa[i], ldfrag(Bp + i * 32, 256, lane), a);
            float bb = pred_b[nt * 16 + (lane & 15)];
#pragma unroll
            for (int r = 0; r < 8; ++r) a[r] += bb;
            st_f32(out + (size_t)b * Ss * NFf + (size_t)mt * 16 * 64 + nt * 16, 64, lane, a);
        }
    }
}

// ---------------- host launcher ----------------
extern "C" void kernel_launch(void* const* d_in, const int* in_sizes, int n_in,
                              void* d_out, int out_size, void* d_ws, size_t ws_size,
                              hipStream_t stream) {
    (void)in_sizes; (void)n_in; (void)out_size; (void)ws_size;
    const float* inputs = (const float*)d_in[0];
    const float* Wq = (const float*)d_in[1];  const float* bq = (const float*)d_in[2];
    const float* Wk = (const float*)d_in[3];  const float* bk = (const float*)d_in[4];
    const float* Wv = (const float*)d_in[5];  const float* bv = (const float*)d_in[6];
    const float* Wo = (const float*)d_in[7];  const float* bo = (const float*)d_in[8];
    const float* W0 = (const float*)d_in[9];  const float* U0 = (const float*)d_in[10]; const float* b0 = (const float*)d_in[11];
    const float* W1 = (const float*)d_in[12]; const float* U1 = (const float*)d_in[13]; const float* b1 = (const float*)d_in[14];
    const float* W2 = (const float*)d_in[15]; const float* U2 = (const float*)d_in[16]; const float* b2 = (const float*)d_in[17];
    const float* Fw0 = (const float*)d_in[18]; const float* Fb0 = (const float*)d_in[19];
    const float* Fw1 = (const float*)d_in[20]; const float* Fb1 = (const float*)d_in[21];
    const float* Fw2 = (const float*)d_in[22]; const float* Fb2 = (const float*)d_in[23];
    const float* attn_scale = (const float*)d_in[24];
    const float* pred_W = (const float*)d_in[25]; const float* pred_b = (const float*)d_in[26];

    char* ws = (char*)d_ws;
    size_t off = 0;
    auto carve = [&](size_t bytes) -> char* {
        char* r = ws + off;
        off = (off + bytes + 255) & ~(size_t)255;
        return r;
    };
    _Float16* inph  = (_Float16*)carve((size_t)Bb * Tt * FAa * 2);
    _Float16* xh    = (_Float16*)carve((size_t)Bb * Tt * FAa * 2);
    _Float16* attn  = (_Float16*)carve((size_t)Bb * Tt * (Hh * KDd) * 2);
    _Float16* pbuf  = (_Float16*)carve((size_t)Bb * Ss * Uu * 2);
    _Float16* WqT   = (_Float16*)carve((size_t)256 * 64 * 2);
    _Float16* WkT   = (_Float16*)carve((size_t)256 * 64 * 2);
    _Float16* WvT   = (_Float16*)carve((size_t)256 * 64 * 2);
    _Float16* WoT   = (_Float16*)carve((size_t)64 * 256 * 2);
    _Float16* W0T   = (_Float16*)carve((size_t)1024 * 64 * 2);
    _Float16* U0T   = (_Float16*)carve((size_t)1024 * 256 * 2);
    _Float16* W1T   = (_Float16*)carve((size_t)1024 * 256 * 2);
    _Float16* U1T   = (_Float16*)carve((size_t)1024 * 256 * 2);
    _Float16* W2T   = (_Float16*)carve((size_t)1024 * 256 * 2);
    _Float16* U2T   = (_Float16*)carve((size_t)1024 * 256 * 2);
    _Float16* F0T   = (_Float16*)carve((size_t)256 * 256 * 2);
    _Float16* F1T   = (_Float16*)carve((size_t)256 * 256 * 2);
    _Float16* F2T   = (_Float16*)carve((size_t)64 * 256 * 2);
    _Float16* pWT   = (_Float16*)carve((size_t)64 * 256 * 2);

    {
        int n = Bb * Tt * FAa;
        k_cvt<<<(n + 255) / 256, 256, 0, stream>>>(inputs, inph, n);
    }
    auto cvtT = [&](const float* w, _Float16* wt, int K, int N) {
        int n = K * N;
        k_cvtT<<<(n + 255) / 256, 256, 0, stream>>>(w, wt, K, N);
    };
    cvtT(Wq, WqT, 64, 256);    cvtT(Wk, WkT, 64, 256);   cvtT(Wv, WvT, 64, 256);
    cvtT(Wo, WoT, 256, 64);
    cvtT(W0, W0T, 64, 1024);   cvtT(U0, U0T, 256, 1024);
    cvtT(W1, W1T, 256, 1024);  cvtT(U1, U1T, 256, 1024);
    cvtT(W2, W2T, 256, 1024);  cvtT(U2, U2T, 256, 1024);
    cvtT(Fw0, F0T, 256, 256);  cvtT(Fw1, F1T, 256, 256); cvtT(Fw2, F2T, 256, 64);
    cvtT(pred_W, pWT, 256, 64);

    constexpr size_t ATTN_LDS = (size_t)3 * 256 * 64 * 2 + (size_t)8 * 16 * 256 * 2;  // 160 KB
    k_attn<<<Bb * Hh, 256, ATTN_LDS, stream>>>(inph, WqT, bq, WkT, bk, WvT, bv, attn);

    k_proj<<<(Bb * Tt / 16) / 8, 256, 0, stream>>>(attn, WoT, bo, xh);

    constexpr size_t LSTM_LDS = 65536 + 3 * (16 * 256 * 2) + 3 * (16 * 256 * 4)
                              + 16 * 64 * 2 + 2 * (16 * 256 * 2);  // ~154 KB
    k_lstm<<<Bb / 16, 256, LSTM_LDS, stream>>>(xh,
        W0T, U0T, b0, W1T, U1T, b1, W2T, U2T, b2,
        F0T, Fb0, F1T, Fb1, F2T, Fb2, pbuf);

    constexpr size_t FIN_LDS = (size_t)64 * 256 * 2 + (size_t)256 * 64 * 2
                             + (size_t)64 * 64 * 4 + (size_t)64 * 64 * 2 + (size_t)64 * 256 * 2;  // 120 KB
    k_final<<<Bb, 256, FIN_LDS, stream>>>(pbuf, attn_scale, pWT, pred_b, (float*)d_out);
}
